// Block_90520730731256
// MI455X (gfx1250) — matrix operand order repaired and checked by reading, not compiled, not run on hardware
//
#include <hip/hip_runtime.h>
#include <hip/hip_bf16.h>
#include <math.h>

// ---------------------------------------------------------------------------
// Types / WMMA helpers (CDNA5 gfx1250, wave32)
// ---------------------------------------------------------------------------
typedef __attribute__((ext_vector_type(16))) _Float16 v16h;
typedef __attribute__((ext_vector_type(8)))  _Float16 v8h;
typedef __attribute__((ext_vector_type(8)))  float    v8f;

#define WMMA_F16(A, B, C) \
    __builtin_amdgcn_wmma_f32_16x16x32_f16(false, (A), false, (B), (short)0, (C), false, false)

// A-matrix 16x32 (MxK) k order used in this file (contiguous; B packed to match):
//   lane: m = lane&15, half = lane>>4
//   elements 0..7  <- k = 16*half + 0..7     (contiguous)
//   elements 8..15 <- k = 16*half + 8..15    (contiguous)
// => two 16B loads from a row-major tile.
__device__ __forceinline__ v16h frag_A_ld(const _Float16* base, int ld, int lane) {
    int m = lane & 15, half = lane >> 4;
    const _Float16* r = base + m * ld + 16 * half;
    v8h lo = *(const v8h*)(r);
    v8h hi = *(const v8h*)(r + 8);
    return __builtin_shufflevector(lo, hi, 0, 1, 2, 3, 4, 5, 6, 7,
                                   8, 9, 10, 11, 12, 13, 14, 15);
}

// B-matrix 32x16 (KxN) order used here (same k order as frag_A_ld): lane n = lane&15, k = 16*(lane>>4) + elem.
// Weights are pre-swizzled so each lane's 16 halves are contiguous (32 B).
__device__ __forceinline__ v16h frag_B_swz(const _Float16* tile, int lane) {
    return *(const v16h*)(tile + (size_t)lane * 16);
}

__device__ __forceinline__ float mp_silu_f(float v) {
    return v / (1.0f + __expf(-v)) * (1.0f / 0.596f);
}

#define RS_RES 1.3130643285972257f   // 1/sqrt(0.7^2+0.3^2)
#define EPSN   1e-4f

// ---------------------------------------------------------------------------
// 1) Weight prep: double mp-normalization folds into one per-row scale.
//    Output is f16 in WMMA-B fragment order:
//      tile (kk, nt) of 32k x 16n -> 512 halves; within tile, lane-major:
//      idx = (kk*ntiles + nt)*512 + ((kl>>4)*16 + (co&15))*16 + (kl&15)
//    conv3 reorders k = (kh,kw,ci) for im2col.
// ---------------------------------------------------------------------------
__global__ void prep_weight(const float* __restrict__ w, _Float16* __restrict__ out,
                            int fanin, int ncols, const float* __restrict__ gain, int conv3) {
    int row = blockIdx.x;                       // output channel co
    const float* wr = w + (size_t)row * fanin;
    float s = 0.f;
    for (int j = threadIdx.x; j < fanin; j += 256) { float v = wr[j]; s += v * v; }
#pragma unroll
    for (int m = 1; m <= 16; m <<= 1) s += __shfl_xor(s, m, 32);
    __shared__ float sred[8];
    __shared__ float sfac;
    int wid = threadIdx.x >> 5, lane = threadIdx.x & 31;
    if (lane == 0) sred[wid] = s;
    __syncthreads();
    if (threadIdx.x == 0) {
        float tot = 0.f;
#pragma unroll
        for (int i = 0; i < 8; ++i) tot += sred[i];
        float sn = rsqrtf((float)fanin);
        float n1 = sqrtf(tot);
        float d1 = EPSN + sn * n1;
        float n2 = n1 / d1;
        float d2 = EPSN + sn * n2;
        float g  = gain ? gain[0] : 1.0f;
        sfac = g * sn / (d1 * d2);
    }
    __syncthreads();
    float fac = sfac;
    int cin = fanin / 9;
    int ntiles = ncols >> 4;
    int nt = row >> 4, nl = row & 15;
    for (int j = threadIdx.x; j < fanin; j += 256) {
        int k = conv3 ? ((j % 9) * cin + (j / 9)) : j;   // w[co][ci][kh][kw] -> (khw,ci)
        int kk = k >> 5, kl = k & 31;
        size_t idx = ((size_t)(kk * ntiles + nt)) * 512
                   + (size_t)((((kl >> 4) * 16 + nl) * 16) + (kl & 15));
        out[idx] = (_Float16)(wr[j] * fac);
    }
}

// ---------------------------------------------------------------------------
// 2) pixel_normalize(x, C=384) + mp_silu; thread = pixel, coalesced over hw.
// ---------------------------------------------------------------------------
__global__ void pixnorm_silu(const float* __restrict__ x, float* __restrict__ xn,
                             _Float16* __restrict__ act0) {
    int p = blockIdx.x * blockDim.x + threadIdx.x;     // 0..16383
    int n = p >> 10, hw = p & 1023;
    const float* xp = x + ((size_t)n * 384) * 1024 + hw;
    float s = 0.f;
    for (int c = 0; c < 384; ++c) { float v = xp[(size_t)c * 1024]; s += v * v; }
    float inv = 1.f / (EPSN + rsqrtf(384.f) * sqrtf(s));
    float* xnp = xn + ((size_t)n * 384) * 1024 + hw;
    _Float16* ap = act0 + (size_t)p * 384;
    for (int c = 0; c < 384; ++c) {
        float v = xp[(size_t)c * 1024] * inv;
        xnp[(size_t)c * 1024] = v;
        ap[c] = (_Float16)mp_silu_f(v);
    }
}

// ---------------------------------------------------------------------------
// 3) emb GEMM: c = emb(16x1536) @ wEmbB(1536x384) + 1 ; one wave, M=16 tile.
// ---------------------------------------------------------------------------
__global__ void emb_gemm_wmma(const float* __restrict__ emb, const _Float16* __restrict__ wB,
                              float* __restrict__ cemb) {
    int lane = threadIdx.x & 31;
    int col = lane & 15, half = lane >> 4;
    int n0 = blockIdx.x * 16;
    v8f acc = {};
    for (int k0 = 0; k0 < 1536; k0 += 32) {
        v16h a;
        const float* r = emb + (size_t)col * 1536 + k0 + 16 * half;  // m = col
#pragma unroll
        for (int i = 0; i < 8; ++i) {
            a[i]     = (_Float16)r[i];
            a[8 + i] = (_Float16)r[8 + i];
        }
        v16h b = frag_B_swz(wB + ((size_t)((k0 >> 5) * 24 + (n0 >> 4))) * 512, lane);
        acc = WMMA_F16(a, b, acc);
    }
#pragma unroll
    for (int r = 0; r < 8; ++r) {
        int m = r + 8 * half;
        cemb[(size_t)m * 384 + n0 + col] = acc[r] + 1.0f;   // +1.0 from reference
    }
}

// ---------------------------------------------------------------------------
// 4) 3x3 conv as implicit GEMM (M=16384 pixels, K=3456, N=384), f16 WMMA.
//    Block: 256 thr (8 waves), tile 128(M) x 64(N); wave = 32x32 (2x2 WMMA).
//    A staged in LDS double-buffered (im2col, zero-pad); B fragments from L2.
//    EPI 0: y *= cemb[n,co]; silu -> act_out (f16 NHWC)
//    EPI 1: mp_sum(xn, y, .3) -> out_f32 (fp32 NCHW) + act_out (f16 NHWC)
// ---------------------------------------------------------------------------
template <int EPI>
__global__ __launch_bounds__(256) void conv3x3_wmma(
        const _Float16* __restrict__ act, const _Float16* __restrict__ wB,
        const float* __restrict__ cemb, const float* __restrict__ xn,
        _Float16* __restrict__ act_out, float* __restrict__ out_f32) {
    __shared__ __align__(32) _Float16 As[2][128 * 32];   // 16 KB, double buffered
    int t = threadIdx.x;
    int wid = t >> 5, lane = t & 31;
    int wm = wid & 3, wn = wid >> 2;
    int bx = blockIdx.x, by = blockIdx.y;
    int nt0 = by * 4 + wn * 2;                        // B tile column (16-wide units)
    v8f acc[2][2] = {};

    // invariant per-thread A-staging coordinates
    int rowA = t >> 1, part = t & 1;
    int pA = bx * 128 + rowA;
    int nA = pA >> 10, hwA = pA & 1023, hA = hwA >> 5, wAx = hwA & 31;
    const _Float16* actb = act + (size_t)nA * 1024 * 384 + part * 16;

    {   // prologue: tile kk=0 (tap dy=dx=-1, c-chunk 0)
        int sh = hA - 1, sw = wAx - 1;
        uint4 d0 = {0, 0, 0, 0}, d1 = {0, 0, 0, 0};
        if (sh >= 0 && sw >= 0) {
            const uint4* src = (const uint4*)(actb + (size_t)(sh * 32 + sw) * 384);
            d0 = src[0]; d1 = src[1];
        }
        uint4* dst = (uint4*)(&As[0][rowA * 32 + part * 16]);
        dst[0] = d0; dst[1] = d1;
    }
    __syncthreads();

    for (int kk = 0; kk < 108; ++kk) {
        // stage next A tile into registers first (latency hides under WMMA)
        uint4 d0 = {0, 0, 0, 0}, d1 = {0, 0, 0, 0};
        bool havnext = (kk + 1 < 108);
        if (havnext) {
            int kn = kk + 1;
            int khw = kn / 12, cc = kn - khw * 12;      // 12 c-chunks per tap
            int dy = khw / 3 - 1, dx = khw - (khw / 3) * 3 - 1;
            int sh = hA + dy, sw = wAx + dx;
            if (sh >= 0 && sh < 32 && sw >= 0 && sw < 32) {
                const uint4* src = (const uint4*)(actb +
                    ((size_t)(sh * 32 + sw) * 384 + cc * 32));
                d0 = src[0]; d1 = src[1];
            }
        }
        const _Float16* btile = wB + ((size_t)kk * 24 + nt0) * 512;
        v16h b0 = frag_B_swz(btile, lane);
        v16h b1 = frag_B_swz(btile + 512, lane);
        const _Float16* Ab = &As[kk & 1][0];
        v16h a0 = frag_A_ld(Ab + (wm * 32) * 32, 32, lane);
        v16h a1 = frag_A_ld(Ab + (wm * 32 + 16) * 32, 32, lane);
        acc[0][0] = WMMA_F16(a0, b0, acc[0][0]);
        acc[0][1] = WMMA_F16(a0, b1, acc[0][1]);
        acc[1][0] = WMMA_F16(a1, b0, acc[1][0]);
        acc[1][1] = WMMA_F16(a1, b1, acc[1][1]);
        if (havnext) {
            uint4* dst = (uint4*)(&As[(kk + 1) & 1][rowA * 32 + part * 16]);
            dst[0] = d0; dst[1] = d1;
        }
        __syncthreads();
    }

    int col = lane & 15, halfL = lane >> 4;
#pragma unroll
    for (int i = 0; i < 2; ++i)
#pragma unroll
        for (int j = 0; j < 2; ++j)
#pragma unroll
            for (int r = 0; r < 8; ++r) {
                int p  = bx * 128 + wm * 32 + i * 16 + r + 8 * halfL;
                int co = by * 64 + wn * 32 + j * 16 + col;
                int n = p >> 10, hw = p & 1023;
                float v = acc[i][j][r];
                if (EPI == 0) {
                    v *= cemb[(size_t)n * 384 + co];
                    act_out[(size_t)p * 384 + co] = (_Float16)mp_silu_f(v);
                } else {
                    float xv = xn[((size_t)n * 384 + co) * 1024 + hw];
                    float r1 = (xv + 0.3f * (v - xv)) * RS_RES;
                    out_f32[((size_t)n * 384 + co) * 1024 + hw] = r1;
                    act_out[(size_t)p * 384 + co] = (_Float16)r1;
                }
            }
}

// ---------------------------------------------------------------------------
// 5) 1x1 conv GEMM (M=16384, K=384, N=1152 or 384), double-buffered A.
//    EPI 0: qkv scatter: q,k -> [nh][s][c]; v -> transposed [nh][c][s]
//    EPI 1: proj: mp_sum(x1, y, .3) + clip -> final fp32 output NCHW
// ---------------------------------------------------------------------------
#define QSZ ((size_t)16 * 6 * 1024 * 64)

template <int EPI>
__global__ __launch_bounds__(256) void gemm1x1_wmma(
        const _Float16* __restrict__ A, const _Float16* __restrict__ B, int ntiles,
        const float* __restrict__ x1, float* __restrict__ out_f32,
        _Float16* __restrict__ out_h) {
    __shared__ __align__(32) _Float16 As[2][128 * 32];
    int t = threadIdx.x;
    int wid = t >> 5, lane = t & 31;
    int wm = wid & 3, wn = wid >> 2;
    int bx = blockIdx.x, by = blockIdx.y;
    int nt0 = by * 4 + wn * 2;
    v8f acc[2][2] = {};

    int rowA = t >> 1, part = t & 1;
    const _Float16* arow = A + (size_t)(bx * 128 + rowA) * 384 + part * 16;

    {   // prologue: tile kk=0
        const uint4* src = (const uint4*)(arow);
        uint4* dst = (uint4*)(&As[0][rowA * 32 + part * 16]);
        dst[0] = src[0]; dst[1] = src[1];
    }
    __syncthreads();

    for (int kk = 0; kk < 12; ++kk) {
        uint4 d0, d1;
        bool havnext = (kk + 1 < 12);
        if (havnext) {
            const uint4* src = (const uint4*)(arow + (kk + 1) * 32);
            d0 = src[0]; d1 = src[1];
        }
        const _Float16* btile = B + ((size_t)kk * ntiles + nt0) * 512;
        v16h b0 = frag_B_swz(btile, lane);
        v16h b1 = frag_B_swz(btile + 512, lane);
        const _Float16* Ab = &As[kk & 1][0];
        v16h a0 = frag_A_ld(Ab + (wm * 32) * 32, 32, lane);
        v16h a1 = frag_A_ld(Ab + (wm * 32 + 16) * 32, 32, lane);
        acc[0][0] = WMMA_F16(a0, b0, acc[0][0]);
        acc[0][1] = WMMA_F16(a0, b1, acc[0][1]);
        acc[1][0] = WMMA_F16(a1, b0, acc[1][0]);
        acc[1][1] = WMMA_F16(a1, b1, acc[1][1]);
        if (havnext) {
            uint4* dst = (uint4*)(&As[(kk + 1) & 1][rowA * 32 + part * 16]);
            dst[0] = d0; dst[1] = d1;
        }
        __syncthreads();
    }

    int col = lane & 15, halfL = lane >> 4;
#pragma unroll
    for (int i = 0; i < 2; ++i)
#pragma unroll
        for (int j = 0; j < 2; ++j)
#pragma unroll
            for (int r = 0; r < 8; ++r) {
                int p  = bx * 128 + wm * 32 + i * 16 + r + 8 * halfL;
                int co = by * 64 + wn * 32 + j * 16 + col;
                int n = p >> 10, hw = p & 1023;
                float v = acc[i][j][r];
                if (EPI == 0) {
                    // channel ch = head*192 + c*3 + which  (reshape N,6,64,3,S)
                    int head = co / 192;
                    int rem  = co - head * 192;
                    int c    = rem / 3;
                    int which = rem - c * 3;
                    size_t nh = (size_t)n * 6 + head;
                    if (which == 2)   // V transposed: [nh][c][s]
                        out_h[2 * QSZ + (nh * 64 + c) * 1024 + hw] = (_Float16)v;
                    else
                        out_h[(size_t)which * QSZ + (nh * 1024 + hw) * 64 + c] = (_Float16)v;
                } else {
                    size_t idx = ((size_t)n * 384 + co) * 1024 + hw;
                    float xv = x1[idx];
                    float o = (xv + 0.3f * (v - xv)) * RS_RES;
                    out_f32[idx] = fminf(fmaxf(o, -256.f), 256.f);
                }
            }
}

// ---------------------------------------------------------------------------
// 6) pixel_normalize over the 64-channel head dim; one wave per vector.
//    q,k: in place ([nh][s][c]); v: in place on transposed layout [nh][c][s].
// ---------------------------------------------------------------------------
__global__ void qkv_norm(_Float16* __restrict__ qkvn) {
    int lane = threadIdx.x & 31;
    int wid = threadIdx.x >> 5;
    size_t vec = (size_t)blockIdx.x * 8 + wid;         // 3*96*1024 vectors
    float a, b, s;
    if (vec < 2 * (size_t)96 * 1024) {                 // q,k: contiguous 64-vector
        _Float16* vp = qkvn + vec * 64 + lane * 2;
        a = (float)vp[0]; b = (float)vp[1];
        s = a * a + b * b;
#pragma unroll
        for (int m = 1; m <= 16; m <<= 1) s += __shfl_xor(s, m, 32);
        float inv = 1.f / (EPSN + 0.125f * sqrtf(s));
        vp[0] = (_Float16)(a * inv);
        vp[1] = (_Float16)(b * inv);
    } else {                                           // v: strided over [c][s]
        size_t nhs = vec - 2 * (size_t)96 * 1024;
        size_t nh = nhs >> 10;
        int sidx = (int)(nhs & 1023);
        _Float16* vp = qkvn + 2 * QSZ + (nh * 64 + lane * 2) * 1024 + sidx;
        a = (float)vp[0]; b = (float)vp[1024];
        s = a * a + b * b;
#pragma unroll
        for (int m = 1; m <= 16; m <<= 1) s += __shfl_xor(s, m, 32);
        float inv = 1.f / (EPSN + 0.125f * sqrtf(s));
        vp[0]    = (_Float16)(a * inv);
        vp[1024] = (_Float16)(b * inv);
    }
}

// ---------------------------------------------------------------------------
// 7) Flash attention: 6 heads, cph=64, S=1024. Block = 4 waves, each wave owns
//    16 queries; per 32-key step: 4 WMMA (Q.K^T) + online softmax + 4 WMMA (P.V)
//    K tile [key][c] and V tile [c][key] give fully vectorized fragment loads.
//    P staging is per-wave private LDS: per-wave DS ordering makes it barrier-free.
// ---------------------------------------------------------------------------
__global__ __launch_bounds__(128) void attn_wmma(const _Float16* __restrict__ qkvn,
                                                 _Float16* __restrict__ attn_out) {
    const _Float16* qn = qkvn;
    const _Float16* kn = qkvn + QSZ;
    const _Float16* vt = qkvn + 2 * QSZ;               // [nh][c][s]
    int nh = blockIdx.y;                 // n*6+head, 0..95
    int qtile = blockIdx.x;              // 0..15 (64 queries each)
    int t = threadIdx.x;
    int wid = t >> 5, lane = t & 31;
    int col = lane & 15, halfL = lane >> 4;
    int qbase = qtile * 64 + wid * 16;

    __shared__ __align__(32) _Float16 Ks[32 * 64];     // [key][c]   4 KB
    __shared__ __align__(32) _Float16 Vst[64 * 32];    // [c][key]   4 KB
    __shared__ __align__(32) _Float16 Ps[4][16 * 32];  // per-wave P 4 KB

    // Q fragments (held for the whole loop): A 16x32 per c-chunk
    const _Float16* qp = qn + ((size_t)nh * 1024 + qbase) * 64;
    v16h aq0 = frag_A_ld(qp, 64, lane);
    v16h aq1 = frag_A_ld(qp + 32, 64, lane);

    v8f o[4] = {};
    float mrow[8], lrow[8];
#pragma unroll
    for (int r = 0; r < 8; ++r) { mrow[r] = -1e30f; lrow[r] = 0.f; }

    for (int kb = 0; kb < 1024; kb += 32) {
        __syncthreads();
        {   // K tile: 32 keys x 64 c, row-major [key][c]
            int row = t >> 2, seg = (t & 3) * 16;
            const uint4* ks = (const uint4*)(kn + ((size_t)nh * 1024 + kb + row) * 64 + seg);
            uint4* kd = (uint4*)(Ks + row * 64 + seg);
            kd[0] = ks[0]; kd[1] = ks[1];
        }
        {   // V tile: 64 c x 32 keys, [c][key] straight from transposed global V
            int c = t >> 1, seg = (t & 1) * 16;
            const uint4* vs = (const uint4*)(vt + ((size_t)nh * 64 + c) * 1024 + kb + seg);
            uint4* vd = (uint4*)(Vst + c * 32 + seg);
            vd[0] = vs[0]; vd[1] = vs[1];
        }
        __syncthreads();

        // S = Q.K^T (two 16x16 fragments over 32 keys), contraction over c=64
        // B frag elements are contiguous in Ks: start = key*64 + chunk*32 + 16*half
        v8f s0 = {}, s1 = {};
        {
            const _Float16* kr0 = Ks + (size_t)col * 64 + halfL * 16;
            s0 = WMMA_F16(aq0, *(const v16h*)(kr0), s0);
            s0 = WMMA_F16(aq1, *(const v16h*)(kr0 + 32), s0);
            const _Float16* kr1 = kr0 + 16 * 64;
            s1 = WMMA_F16(aq0, *(const v16h*)(kr1), s1);
            s1 = WMMA_F16(aq1, *(const v16h*)(kr1 + 32), s1);
        }

        // online softmax (rows live in vgpr index + half; cols across 16 lanes)
#pragma unroll
        for (int r = 0; r < 8; ++r) {
            float a = s0[r] * 0.125f, b = s1[r] * 0.125f;    // scale = 1/sqrt(64)
            float mx = fmaxf(a, b);
            mx = fmaxf(mx, __shfl_xor(mx, 1, 32));
            mx = fmaxf(mx, __shfl_xor(mx, 2, 32));
            mx = fmaxf(mx, __shfl_xor(mx, 4, 32));
            mx = fmaxf(mx, __shfl_xor(mx, 8, 32));
            float nm = fmaxf(mrow[r], mx);
            float corr = __expf(mrow[r] - nm);
            float p0 = __expf(a - nm);
            float p1 = __expf(b - nm);
            float rs = p0 + p1;
            rs += __shfl_xor(rs, 1, 32);
            rs += __shfl_xor(rs, 2, 32);
            rs += __shfl_xor(rs, 4, 32);
            rs += __shfl_xor(rs, 8, 32);
            lrow[r] = lrow[r] * corr + rs;
            mrow[r] = nm;
#pragma unroll
            for (int cg = 0; cg < 4; ++cg) o[cg][r] *= corr;
            s0[r] = p0; s1[r] = p1;
        }

        // stage P (C layout -> row-major 16x32); same-wave DS ops are in-order,
        // and this region is private to the wave -> no barrier needed
        _Float16* P = &Ps[wid][0];
#pragma unroll
        for (int r = 0; r < 8; ++r) {
            int row = r + 8 * halfL;
            P[row * 32 + col]      = (_Float16)s0[r];
            P[row * 32 + 16 + col] = (_Float16)s1[r];
        }

        v16h ap = frag_A_ld(P, 32, lane);
#pragma unroll
        for (int cg = 0; cg < 4; ++cg) {
            // V frag (B[key][c]) from [c][key] tile: contiguous per lane
            const _Float16* vp = Vst + (size_t)(cg * 16 + col) * 32 + halfL * 16;
            o[cg] = WMMA_F16(ap, *(const v16h*)(vp), o[cg]);
        }
    }

    // write out: (N, head*64+c, H, W) as f16 NHWC for the proj GEMM
    int n = nh / 6, h = nh - n * 6;
#pragma unroll
    for (int r = 0; r < 8; ++r) {
        int srow = qbase + r + 8 * halfL;
        size_t p = (size_t)n * 1024 + srow;
        float inv = 1.f / lrow[r];
#pragma unroll
        for (int cg = 0; cg < 4; ++cg)
            attn_out[p * 384 + h * 64 + cg * 16 + col] = (_Float16)(o[cg][r] * inv);
    }
}

// ---------------------------------------------------------------------------
// Host launcher
// ---------------------------------------------------------------------------
extern "C" void kernel_launch(void* const* d_in, const int* in_sizes, int n_in,
                              void* d_out, int out_size, void* d_ws, size_t ws_size,
                              hipStream_t stream) {
    (void)in_sizes; (void)n_in; (void)out_size; (void)ws_size;
    const float* x        = (const float*)d_in[0];
    const float* emb      = (const float*)d_in[1];
    const float* w_res0   = (const float*)d_in[2];
    const float* w_emb    = (const float*)d_in[3];
    const float* emb_gain = (const float*)d_in[4];
    const float* w_res1   = (const float*)d_in[5];
    const float* w_qkv    = (const float*)d_in[6];
    const float* w_proj   = (const float*)d_in[7];
    float* out = (float*)d_out;

    char* p = (char*)d_ws;
    auto alloc = [&](size_t bytes) {
        char* r = p;
        p += (bytes + 255) & ~(size_t)255;
        return r;
    };
    _Float16* wbR0 = (_Float16*)alloc((size_t)3456 * 384 * 2);
    _Float16* wbR1 = (_Float16*)alloc((size_t)3456 * 384 * 2);
    _Float16* wbE  = (_Float16*)alloc((size_t)1536 * 384 * 2);
    _Float16* wbQ  = (_Float16*)alloc((size_t)384 * 1152 * 2);
    _Float16* wbP  = (_Float16*)alloc((size_t)384 * 384 * 2);
    float*    xn      = (float*)alloc((size_t)16 * 384 * 1024 * 4);
    _Float16* act0    = (_Float16*)alloc((size_t)16384 * 384 * 2);
    float*    cemb    = (float*)alloc((size_t)16 * 384 * 4);
    _Float16* act1    = (_Float16*)alloc((size_t)16384 * 384 * 2);
    float*    x1      = (float*)alloc((size_t)16 * 384 * 1024 * 4);
    _Float16* x1act   = (_Float16*)alloc((size_t)16384 * 384 * 2);
    _Float16* qkvn    = (_Float16*)alloc(3 * QSZ * 2);
    _Float16* attnout = (_Float16*)alloc((size_t)16384 * 384 * 2);

    // weight normalization -> f16, WMMA-fragment-swizzled B layout
    prep_weight<<<384, 256, 0, stream>>>(w_res0, wbR0, 3456, 384, nullptr, 1);
    prep_weight<<<384, 256, 0, stream>>>(w_res1, wbR1, 3456, 384, nullptr, 1);
    prep_weight<<<384, 256, 0, stream>>>(w_emb, wbE, 1536, 384, emb_gain, 0);
    prep_weight<<<1152, 256, 0, stream>>>(w_qkv, wbQ, 384, 1152, nullptr, 0);
    prep_weight<<<384, 256, 0, stream>>>(w_proj, wbP, 384, 384, nullptr, 0);

    // pixel-norm + silu
    pixnorm_silu<<<64, 256, 0, stream>>>(x, xn, act0);

    // emb modulation GEMM (+1)
    emb_gemm_wmma<<<24, 32, 0, stream>>>(emb, wbE, cemb);

    // conv0: silu(y * c) -> act1
    conv3x3_wmma<0><<<dim3(128, 6), 256, 0, stream>>>(act0, wbR0, cemb, nullptr, act1, nullptr);
    // conv1: mp_sum(xn, y) -> x1 (fp32) + x1act (f16)
    conv3x3_wmma<1><<<dim3(128, 6), 256, 0, stream>>>(act1, wbR1, nullptr, xn, x1act, x1);

    // qkv projection, scattered to attention layout (V transposed)
    gemm1x1_wmma<0><<<dim3(128, 18), 256, 0, stream>>>(x1act, wbQ, 72, nullptr, nullptr, qkvn);
    // per-head channel normalization (in place)
    qkv_norm<<<36864, 256, 0, stream>>>(qkvn);
    // attention
    attn_wmma<<<dim3(16, 96), 128, 0, stream>>>(qkvn, attnout);
    // proj + final residual + clip
    gemm1x1_wmma<1><<<dim3(128, 6), 256, 0, stream>>>(attnout, wbP, 24, x1, out, nullptr);
}
